// DecoderPolicyGradient_30339648979345
// MI455X (gfx1250) — compile-verified, matrix operand order checked by reading, not run on hardware
//
#include <hip/hip_runtime.h>
#include <hip/hip_bf16.h>
#include <math.h>

// Problem constants (fixed by the reference).
#define B_ 128
#define T_ 20
#define E_ 512
#define H_ 512
#define V_ 10000

typedef __attribute__((ext_vector_type(16))) __bf16 v16bf;
typedef __attribute__((ext_vector_type(8)))  float  v8f;
typedef __attribute__((ext_vector_type(4)))  unsigned int v4u;

union BfFrag { v16bf bf; v4u u[2]; };

__device__ __forceinline__ unsigned short f2bf(float x) {
    unsigned int u = __float_as_uint(x);
    u += 0x7FFFu + ((u >> 16) & 1u);   // round-to-nearest-even
    return (unsigned short)(u >> 16);
}

// ---------------------------------------------------------------------------
// Build X (time-major [T*B, E], bf16): row 0..B-1 = features, rows for t>=1 =
// embed_w[captions[:, t-1]].
// ---------------------------------------------------------------------------
__global__ void prep_x_kernel(const float* __restrict__ features,
                              const int* __restrict__ captions,
                              const float* __restrict__ embed_w,
                              unsigned short* __restrict__ Xbf) {
    int idx = blockIdx.x * blockDim.x + threadIdx.x;
    const int total = T_ * B_ * E_;
    if (idx >= total) return;
    int e   = idx % E_;
    int row = idx / E_;
    int b   = row % B_;
    int t   = row / B_;
    float v;
    if (t == 0) {
        v = features[(size_t)b * E_ + e];
    } else {
        int tok = captions[b * T_ + (t - 1)];
        v = embed_w[(size_t)tok * E_ + e];
    }
    Xbf[idx] = f2bf(v);
}

__global__ void f32_to_bf16_kernel(const float* __restrict__ src,
                                   unsigned short* __restrict__ dst, int n) {
    int i = blockIdx.x * blockDim.x + threadIdx.x;
    if (i < n) dst[i] = f2bf(src[i]);
}

// ---------------------------------------------------------------------------
// C[M,N] (f32) = A[M,K](bf16) @ Bw[N,K](bf16)^T + bias0 + bias1
// One wave computes a 16 x (16*NT) strip via v_wmma_f32_16x16x32_bf16.
// ---------------------------------------------------------------------------
template<int NT>
__global__ __launch_bounds__(256)
void gemm_bf16_wmma_kernel(const unsigned short* __restrict__ A,
                           const unsigned short* __restrict__ Bw,
                           const float* __restrict__ bias0,
                           const float* __restrict__ bias1,
                           float* __restrict__ C,
                           int M, int N, int K) {
    const int lane = threadIdx.x & 31;
    const int wave = (blockIdx.x * blockDim.x + threadIdx.x) >> 5;
    const int nStrips = N / (16 * NT);
    const int mt = wave / nStrips;
    const int ns = wave % nStrips;
    if (mt * 16 >= M) return;

    const int half = lane >> 4;   // which 16-lane half
    const int nl   = lane & 15;   // row (A) / col (B) within tile

    const unsigned short* aRow = A + (size_t)(mt * 16 + nl) * K;

    v8f acc[NT] = {};

    const int kSteps = K >> 5;    // K blocks of 32
    for (int kk = 0; kk < kSteps; ++kk) {
        const int k0 = kk * 32 + half * 8;
        BfFrag a;
        a.u[0] = *reinterpret_cast<const v4u*>(aRow + k0);
        a.u[1] = *reinterpret_cast<const v4u*>(aRow + k0 + 16);
        #pragma unroll
        for (int j = 0; j < NT; ++j) {
            const unsigned short* bRow =
                Bw + (size_t)((ns * NT + j) * 16 + nl) * K;
            BfFrag b;
            b.u[0] = *reinterpret_cast<const v4u*>(bRow + k0);
            b.u[1] = *reinterpret_cast<const v4u*>(bRow + k0 + 16);
            acc[j] = __builtin_amdgcn_wmma_f32_16x16x32_bf16(
                false, a.bf, false, b.bf, (short)0, acc[j], false, false);
        }
    }

    #pragma unroll
    for (int j = 0; j < NT; ++j) {
        const int nb = (ns * NT + j) * 16 + nl;
        float bias = 0.0f;
        if (bias0) bias += bias0[nb];
        if (bias1) bias += bias1[nb];
        #pragma unroll
        for (int r = 0; r < 8; ++r) {
            const int m = mt * 16 + r + 8 * half;
            C[(size_t)m * N + nb] = acc[j][r] + bias;
        }
    }
}

// ---------------------------------------------------------------------------
// One LSTM timestep, fused: gates = Gx[t] + h_{t-1} @ W_hh^T, then the cell.
// Each wave owns a 16-row x 16-col chunk of the hidden state and computes the
// matching 16x16 tile of all four gates (i,f,g,o) so the nonlinearity can be
// applied in-register. 8 M-tiles x 32 col-groups = 256 waves.
// ---------------------------------------------------------------------------
__global__ __launch_bounds__(256)
void lstm_step_kernel(const unsigned short* __restrict__ Hprev, // [B,H] bf16
                      unsigned short* __restrict__ Hout,        // [B,H] bf16
                      const unsigned short* __restrict__ Whh,   // [4H,H] bf16
                      const float* __restrict__ Gx,             // [B,4H] f32 (biases folded)
                      float* __restrict__ Cst) {                // [B,H] f32 in/out
    const int lane = threadIdx.x & 31;
    const int wave = (blockIdx.x * blockDim.x + threadIdx.x) >> 5;
    const int mt = wave >> 5;      // 0..7
    const int ng = wave & 31;      // 0..31
    if (mt >= B_ / 16) return;

    const int half = lane >> 4;
    const int nl   = lane & 15;
    const int col  = ng * 16 + nl;

    // Init accumulators with the precomputed input-side gates.
    v8f acc[4] = {};
    #pragma unroll
    for (int g = 0; g < 4; ++g) {
        #pragma unroll
        for (int r = 0; r < 8; ++r) {
            const int m = mt * 16 + r + 8 * half;
            acc[g][r] = Gx[(size_t)m * (4 * H_) + g * H_ + col];
        }
    }

    const unsigned short* aRow = Hprev + (size_t)(mt * 16 + nl) * H_;
    for (int kk = 0; kk < (H_ >> 5); ++kk) {
        const int k0 = kk * 32 + half * 8;
        BfFrag a;
        a.u[0] = *reinterpret_cast<const v4u*>(aRow + k0);
        a.u[1] = *reinterpret_cast<const v4u*>(aRow + k0 + 16);
        #pragma unroll
        for (int g = 0; g < 4; ++g) {
            const unsigned short* bRow = Whh + (size_t)(g * H_ + col) * H_;
            BfFrag b;
            b.u[0] = *reinterpret_cast<const v4u*>(bRow + k0);
            b.u[1] = *reinterpret_cast<const v4u*>(bRow + k0 + 16);
            acc[g] = __builtin_amdgcn_wmma_f32_16x16x32_bf16(
                false, a.bf, false, b.bf, (short)0, acc[g], false, false);
        }
    }

    #pragma unroll
    for (int r = 0; r < 8; ++r) {
        const int m = mt * 16 + r + 8 * half;
        const float ig = 1.0f / (1.0f + expf(-acc[0][r]));
        const float fg = 1.0f / (1.0f + expf(-acc[1][r]));
        const float gg = tanhf(acc[2][r]);
        const float og = 1.0f / (1.0f + expf(-acc[3][r]));
        const size_t ci = (size_t)m * H_ + col;
        const float c = fg * Cst[ci] + ig * gg;
        Cst[ci] = c;
        Hout[ci] = f2bf(og * tanhf(c));
    }
}

// ---------------------------------------------------------------------------
extern "C" void kernel_launch(void* const* d_in, const int* in_sizes, int n_in,
                              void* d_out, int out_size, void* d_ws, size_t ws_size,
                              hipStream_t stream) {
    (void)in_sizes; (void)n_in; (void)out_size; (void)ws_size;

    const float* features = (const float*)d_in[0];
    const int*   captions = (const int*)d_in[1];
    const float* h0       = (const float*)d_in[2];
    const float* c0       = (const float*)d_in[3];
    const float* embed_w  = (const float*)d_in[4];
    const float* W_ih     = (const float*)d_in[5];
    const float* W_hh     = (const float*)d_in[6];
    const float* b_ih     = (const float*)d_in[7];
    const float* b_hh     = (const float*)d_in[8];
    const float* W_lin    = (const float*)d_in[9];
    const float* b_lin    = (const float*)d_in[10];

    // Workspace carve-up (256B aligned blocks), ~41 MB total.
    char* ws = (char*)d_ws;
    size_t off = 0;
    auto alloc = [&](size_t bytes) -> char* {
        char* p = ws + off;
        off += (bytes + 255) & ~(size_t)255;
        return p;
    };
    unsigned short* Xbf   = (unsigned short*)alloc((size_t)T_ * B_ * E_ * 2);       // 2.62 MB
    unsigned short* WihB  = (unsigned short*)alloc((size_t)4 * H_ * E_ * 2);        // 2.10 MB
    unsigned short* WhhB  = (unsigned short*)alloc((size_t)4 * H_ * H_ * 2);        // 2.10 MB
    unsigned short* WlinB = (unsigned short*)alloc((size_t)V_ * H_ * 2);            // 10.24 MB
    float*          Gx    = (float*)alloc((size_t)T_ * B_ * 4 * H_ * 4);            // 20.97 MB
    unsigned short* Hall  = (unsigned short*)alloc((size_t)(T_ + 1) * B_ * H_ * 2); // 2.75 MB
    float*          Cst   = (float*)alloc((size_t)B_ * H_ * 4);                     // 0.26 MB

    // 1) Build X (embedding gather + feature row), convert weights to bf16.
    {
        int total = T_ * B_ * E_;
        prep_x_kernel<<<(total + 255) / 256, 256, 0, stream>>>(features, captions, embed_w, Xbf);
    }
    f32_to_bf16_kernel<<<(4 * H_ * E_ + 255) / 256, 256, 0, stream>>>(W_ih, WihB, 4 * H_ * E_);
    f32_to_bf16_kernel<<<(4 * H_ * H_ + 255) / 256, 256, 0, stream>>>(W_hh, WhhB, 4 * H_ * H_);
    f32_to_bf16_kernel<<<(V_ * H_ + 255) / 256, 256, 0, stream>>>(W_lin, WlinB, V_ * H_);
    f32_to_bf16_kernel<<<(B_ * H_ + 255) / 256, 256, 0, stream>>>(h0, Hall, B_ * H_); // slot 0 = h0
    hipMemcpyAsync(Cst, c0, (size_t)B_ * H_ * sizeof(float), hipMemcpyDeviceToDevice, stream);

    // 2) Gx[T*B, 4H] = X @ W_ih^T + (b_ih + b_hh)   (hoisted out of the recurrence)
    {
        const int M = T_ * B_, N = 4 * H_, K = E_;      // 2560 x 2048 x 512
        const int waves = (M / 16) * (N / 64);          // NT=4 -> 5120 waves
        gemm_bf16_wmma_kernel<4><<<waves * 32 / 256, 256, 0, stream>>>(
            Xbf, WihB, b_ih, b_hh, Gx, M, N, K);
    }

    // 3) Sequential recurrence: 20 fused GEMM+cell steps (256 waves each).
    for (int t = 0; t < T_; ++t) {
        lstm_step_kernel<<<32, 256, 0, stream>>>(
            Hall + (size_t)t * B_ * H_,
            Hall + (size_t)(t + 1) * B_ * H_,
            WhhB,
            Gx + (size_t)t * B_ * 4 * H_,
            Cst);
    }

    // 4) logits[T*B, V] = H @ W_lin^T + b_lin  (time-major rows == output layout)
    {
        const int M = T_ * B_, N = V_, K = H_;          // 2560 x 10000 x 512
        const int waves = (M / 16) * (N / 80);          // NT=5 -> 20000 waves
        gemm_bf16_wmma_kernel<5><<<waves * 32 / 256, 256, 0, stream>>>(
            Hall + (size_t)B_ * H_, WlinB, b_lin, nullptr, (float*)d_out, M, N, K);
    }
}
